// DepthwiseCrossCorrelation_2035814498855
// MI455X (gfx1250) — matrix operand order repaired and checked by reading, not compiled, not run on hardware
//
#include <hip/hip_runtime.h>

// Depthwise cross-correlation, SiamFC style.
// z: [128,256,7,7] f32 template, x: [128,256,31,31] f32 search
// out: [128,256,25,25] f32 response.
//
// Per channel: O = sum_{kr=0..6} X_rows(shift kr) [16x32] * Toeplitz(Z row kr) [32x16]
// using v_wmma_f32_16x16x32_f16, f32 accumulation.

typedef _Float16 v16h __attribute__((ext_vector_type(16)));
typedef _Float16 v8h  __attribute__((ext_vector_type(8)));
typedef float    v8f  __attribute__((ext_vector_type(8)));

#define HZ 7
#define WZ 7
#define HX 31
#define WX 31
#define HO 25
#define WO 25
#define XLDS_STRIDE 32   // 31 cols + 1 zero pad -> K=32
#define NTILES 14        // 7 kernel rows * 2 column-tiles

__global__ __launch_bounds__(128)
void dwxcorr_wmma_kernel(const float* __restrict__ z,
                         const float* __restrict__ x,
                         float* __restrict__ out)
{
    // One block per (b,c) channel slice.
    const int bc   = blockIdx.x;
    const int tid  = threadIdx.x;

    const float* xg = x + (long)bc * (HX * WX);
    const float* zg = z + (long)bc * (HZ * WZ);
    float*       og = out + (long)bc * (HO * WO);

    __shared__ __attribute__((aligned(64))) _Float16 ldsX[HX * XLDS_STRIDE];     // 1984 B
    __shared__ __attribute__((aligned(64))) _Float16 ldsZ[HZ * 8];               // 112 B
    __shared__ __attribute__((aligned(64))) _Float16 ldsB[NTILES][16][32];       // 14 KB, [tile][n][k]

    // ---- Phase 1: stage X and Z into LDS as f16 (coalesced f32 loads) ----
    for (int i = tid; i < HX * WX; i += 128) {
        int r = i / WX, c = i - r * WX;
        ldsX[r * XLDS_STRIDE + c] = (_Float16)xg[i];
    }
    if (tid < HX) ldsX[tid * XLDS_STRIDE + (XLDS_STRIDE - 1)] = (_Float16)0.f; // zero pad col 31
    if (tid < HZ * WZ) {
        int r = tid / WZ, c = tid - r * WZ;
        ldsZ[r * 8 + c] = (_Float16)zg[tid];
    }
    __syncthreads();

    // ---- Phase 2: build Toeplitz B tiles, transposed [n][k] for contiguous-K lane loads ----
    // B_{kr,tc}[k][n] = Z[kr][k - (tc*16 + n)] if 0 <= k-(tc*16+n) <= 6 else 0
    for (int i = tid; i < NTILES * 16 * 32; i += 128) {
        int tile = i >> 9;
        int rem  = i & 511;
        int n    = rem >> 5;
        int k    = rem & 31;
        int kr   = tile >> 1;
        int tc   = tile & 1;
        int t    = k - tc * 16 - n;
        _Float16 v = (t >= 0 && t < WZ) ? ldsZ[kr * 8 + t] : (_Float16)0.f;
        ldsB[tile][n][k] = v;
    }
    __syncthreads();

    // ---- Phase 3: one wave per 16x16 output tile (2x2 tiles cover 25x25) ----
    const int wave   = tid >> 5;        // 0..3
    const int lane   = tid & 31;
    const int laneLo = lane & 15;
    const int laneHi = (lane >> 4) & 1; // upper half-wave
    const int tr     = wave >> 1;       // output row tile
    const int tc     = wave & 1;        // output col tile

    v8f acc = {};

    #pragma unroll
    for (int kr = 0; kr < HZ; ++kr) {
        // A k order used here (contiguous, not the ISA table order): lane<16 holds M=lane, K 0-15;
        // lane>=16 holds M=lane-16, K 16-31.
        int xr = tr * 16 + laneLo + kr;         // X row feeding output row m=laneLo
        if (xr > HX - 1) xr = HX - 1;           // clamp; clamped rows map to invalid outputs only
        const _Float16* xrow = &ldsX[xr * XLDS_STRIDE];

        union { v16h v; v8h h[2]; } a;
        a.h[0] = *(const v8h*)&xrow[laneHi << 4];          // K = laneHi*16 .. +7
        a.h[1] = *(const v8h*)&xrow[(laneHi << 4) + 8];    // K = laneHi*16 + 8 .. +15

        // B k order used here (contiguous, matches A): lane<16 -> N=lane, K=0..15; lane>=16 -> K=16..31.
        // ldsB is stored [n][k], so each lane reads 16 contiguous f16 (32B aligned).
        v16h b = *(const v16h*)&ldsB[kr * 2 + tc][laneLo][laneHi << 4];

        acc = __builtin_amdgcn_wmma_f32_16x16x32_f16(
                  /*neg_a=*/false, a.v,
                  /*neg_b=*/false, b,
                  /*c_mod=*/(short)0, acc,
                  /*reuse_a=*/false, /*reuse_b=*/false);
    }

    // ---- Store: C/D layout: VGPR r -> M = r + 8*laneHi, N = laneLo ----
    const int ocol = tc * 16 + laneLo;
    #pragma unroll
    for (int r = 0; r < 8; ++r) {
        int orow = tr * 16 + r + (laneHi << 3);
        if (orow < HO && ocol < WO)
            og[orow * WO + ocol] = acc[r];
    }
}

extern "C" void kernel_launch(void* const* d_in, const int* in_sizes, int n_in,
                              void* d_out, int out_size, void* d_ws, size_t ws_size,
                              hipStream_t stream)
{
    const float* z = (const float*)d_in[0];   // [128,256,7,7]
    const float* x = (const float*)d_in[1];   // [128,256,31,31]
    float* out = (float*)d_out;               // [128,256,25,25]

    const int channels = 128 * 256;           // one block per channel slice
    dwxcorr_wmma_kernel<<<channels, 128, 0, stream>>>(z, x, out);
}
